// GraphConvolution_31585189495294
// MI455X (gfx1250) — matrix operand order repaired and checked by reading, not compiled, not run on hardware
//
#include <hip/hip_runtime.h>

#define N_NODES 100000
#define N_EDGES 1600000
#define IN_DIM  512
#define OUT_DIM 128

typedef __attribute__((ext_vector_type(16))) __bf16 v16bf;
typedef __attribute__((ext_vector_type(8)))  float  v8f;
typedef __attribute__((ext_vector_type(4)))  unsigned int v4u;
typedef __attribute__((ext_vector_type(8)))  int v8i;
typedef __attribute__((ext_vector_type(4)))  int v4i;

// ---------------- ws layout ----------------
// [0, 51,200,000)            : weighted fp32 [N_NODES][OUT_DIM]
// [51,200,000, +131,072)     : Wt bf16 [OUT_DIM][IN_DIM]   (transposed, K-contiguous)
#define WS_WT_OFF ((size_t)N_NODES * OUT_DIM * sizeof(float))

// ---------------------------------------------------------------------------
// Kernel 1: W fp32 [K=512][N=128]  ->  Wt bf16 [N=128][K=512] (transpose+convert)
// ---------------------------------------------------------------------------
__global__ void k_wt_convert(const float* __restrict__ W, __bf16* __restrict__ Wt) {
    int i = blockIdx.x * blockDim.x + threadIdx.x;   // 0 .. 65535
    int k = i >> 7;          // / OUT_DIM
    int n = i & (OUT_DIM - 1);
    Wt[(size_t)n * IN_DIM + k] = (__bf16)W[(size_t)k * OUT_DIM + n];
}

// ---------------------------------------------------------------------------
// Kernel 2: zero the output accumulator (d_out is poisoned by the harness)
// ---------------------------------------------------------------------------
__global__ void k_zero(float4* __restrict__ out, int n4) {
    int i = blockIdx.x * blockDim.x + threadIdx.x;
    if (i < n4) out[i] = make_float4(0.f, 0.f, 0.f, 0.f);
}

// ---------------------------------------------------------------------------
// Kernel 3: weighted = x @ W   via v_wmma_f32_16x16x32_bf16
//   block = 256 threads (8 waves), block tile 64(M) x 128(N)
//   wave (wm = w&3, wn = w>>2) computes 16(M) x 64(N) = 4 WMMA tiles
//   B tile staged by the Tensor Data Mover each K-step (HW-padded LDS layout);
//   A tile staged by all threads with inline fp32->bf16 conversion.
// LDS rows padded to 80B (20 banks) -> conflict-free, 16B-aligned b128 access.
// ---------------------------------------------------------------------------
#define A_STRIDE 40   // bf16 elements per padded row (64B data + 16B pad)
#define B_STRIDE 40

__launch_bounds__(256, 2)
__global__ void k_gemm(const float* __restrict__ x,
                       const __bf16* __restrict__ Wt,
                       float* __restrict__ wtd) {
    __shared__ __bf16 As[64 * A_STRIDE];    //  5120 B
    __shared__ __bf16 Bs[128 * B_STRIDE];   // 10240 B

    const int tid  = threadIdx.x;
    const int wave = tid >> 5;
    const int lane = tid & 31;
    const int wm   = wave & 3;     // M sub-tile (16 rows each)
    const int wn   = wave >> 2;    // N half (64 cols each)
    const int half = lane >> 4;    // K-half selector per ISA layout
    const int l16  = lane & 15;
    const int row0 = blockIdx.x * 64;

    v8f acc[4] = {};

    // A staging assignment: 64 rows x 32 floats / 256 threads = 8 floats each
    const int arow = tid >> 2;
    const int aks  = (tid & 3) * 8;
    const int grow = row0 + arow;

    // TDM descriptor constants (group1): workgroup_mask=0, data_size=1 (2B),
    // pad_enable=1, pad_interval=3 (16 DWORDs = 64B), pad_amount=3 (4 DWORDs = 16B),
    // tensor_dim0=512, tensor_dim1=128, tile_dim0=32, tile_dim1=128,
    // tensor_dim0_stride=512.
    const unsigned int ldsB = (unsigned int)(uintptr_t)(void*)&Bs[0]; // flat LDS addr[31:0]

    for (int kk = 0; kk < IN_DIM / 32; ++kk) {
        const int k0 = kk * 32;
        __syncthreads();

        // ---- stage B via Tensor Data Mover (one wave issues; EXEC ignored) ----
        if (tid == 0) {
            unsigned long long ga =
                (unsigned long long)(uintptr_t)Wt + (unsigned long long)k0 * 2ull;
            v4u g0;
            g0.x = 1u;                                   // count=1, user descriptor
            g0.y = ldsB;                                 // lds_addr
            g0.z = (unsigned int)ga;                     // global_addr[31:0]
            g0.w = (unsigned int)(ga >> 32) | (2u << 30); // global_addr[56:32] | type=2
            v8i g1 = { 0x06D10000,          // data_size=1|pad_enable|pad_int=3|pad_amt=3
                       0x02000000,          // tensor_dim0 = 512 (low16 << 16)
                       0x00800000,          // tensor_dim1 = 128 (low16 << 16)
                       0x00200000,          // tile_dim0  = 32  (<< 16)
                       128,                 // tile_dim1  = 128
                       512,                 // tensor_dim0_stride = 512
                       0, 0 };
            v4i g2 = { 0, 0, 0, 0 };
            v4i g3 = { 0, 0, 0, 0 };
            v8i g4 = { 0, 0, 0, 0, 0, 0, 0, 0 };
            __builtin_amdgcn_tensor_load_to_lds(g0, g1, g2, g3, g4, 0);
        }

        // ---- stage A: fp32 -> bf16 into LDS (overlaps with the TDM DMA) ----
        {
            union { __bf16 h[8]; uint4 q; } u;
            if (grow < N_NODES) {
                const float* p = x + (size_t)grow * IN_DIM + k0 + aks;
                float4 f0 = *(const float4*)p;
                float4 f1 = *(const float4*)(p + 4);
                u.h[0] = (__bf16)f0.x; u.h[1] = (__bf16)f0.y;
                u.h[2] = (__bf16)f0.z; u.h[3] = (__bf16)f0.w;
                u.h[4] = (__bf16)f1.x; u.h[5] = (__bf16)f1.y;
                u.h[6] = (__bf16)f1.z; u.h[7] = (__bf16)f1.w;
            } else {
                u.q = make_uint4(0u, 0u, 0u, 0u);
            }
            *(uint4*)&As[arow * A_STRIDE + aks] = u.q;
        }

        if (tid == 0) {
            __builtin_amdgcn_s_wait_tensorcnt(0);        // B tile landed in LDS
        }
        __syncthreads();

        // ---- compute: A fragment (16x32), then 4 B fragments, then 4 WMMAs ----
        union frag { v16bf v; uint4 q[2]; };
        frag af;
        const int mrow = wm * 16 + l16;
        af.q[0] = *(const uint4*)&As[mrow * A_STRIDE + half * 8];        // K = half*8 ..
        af.q[1] = *(const uint4*)&As[mrow * A_STRIDE + 16 + half * 8];   // K = 16+half*8 ..

        frag bfr[4];
        #pragma unroll
        for (int t = 0; t < 4; ++t) {
            const int col = wn * 64 + t * 16 + l16;
            bfr[t].q[0] = *(const uint4*)&Bs[col * B_STRIDE + half * 8];
            bfr[t].q[1] = *(const uint4*)&Bs[col * B_STRIDE + 16 + half * 8];
        }
        #pragma unroll
        for (int t = 0; t < 4; ++t) {
            acc[t] = __builtin_amdgcn_wmma_f32_16x16x32_bf16(
                false, af.v, false, bfr[t].v, (short)0, acc[t], false, false);
        }
    }

    // ---- epilogue: C/D layout -> weighted[row][col] fp32 ----
    const int rbase = row0 + wm * 16 + half * 8;
    if (row0 + 64 <= N_NODES) {
        // fast path: whole block in range (1562 of 1563 blocks)
        #pragma unroll
        for (int t = 0; t < 4; ++t) {
            const int col = wn * 64 + t * 16 + l16;
            #pragma unroll
            for (int r = 0; r < 8; ++r)
                wtd[(size_t)(rbase + r) * OUT_DIM + col] = acc[t][r];
        }
    } else {
        #pragma unroll
        for (int t = 0; t < 4; ++t) {
            const int col = wn * 64 + t * 16 + l16;
            #pragma unroll
            for (int r = 0; r < 8; ++r)
                if (rbase + r < N_NODES)
                    wtd[(size_t)(rbase + r) * OUT_DIM + col] = acc[t][r];
        }
    }
}

// ---------------------------------------------------------------------------
// Kernel 4: SpMM scatter: out[dst] += val * weighted[src]
//   one wave32 per edge, 4 dims per lane (float4 gather from L2-resident
//   weighted, 4 global_atomic_add_f32 without return)
// ---------------------------------------------------------------------------
__launch_bounds__(256)
__global__ void k_spmm(const float* __restrict__ wtd,
                       const int*   __restrict__ src,
                       const int*   __restrict__ dst,
                       const float* __restrict__ vals,
                       float* __restrict__ out) {
    const int e    = (int)((blockIdx.x * blockDim.x + threadIdx.x) >> 5);
    const int lane = threadIdx.x & 31;
    if (e >= N_EDGES) return;
    const int   s = src[e];
    const int   d = dst[e];
    const float v = vals[e];
    const float4 m = *(const float4*)(wtd + (size_t)s * OUT_DIM + lane * 4);
    float* o = out + (size_t)d * OUT_DIM + lane * 4;
    __hip_atomic_fetch_add(o + 0, v * m.x, __ATOMIC_RELAXED, __HIP_MEMORY_SCOPE_AGENT);
    __hip_atomic_fetch_add(o + 1, v * m.y, __ATOMIC_RELAXED, __HIP_MEMORY_SCOPE_AGENT);
    __hip_atomic_fetch_add(o + 2, v * m.z, __ATOMIC_RELAXED, __HIP_MEMORY_SCOPE_AGENT);
    __hip_atomic_fetch_add(o + 3, v * m.w, __ATOMIC_RELAXED, __HIP_MEMORY_SCOPE_AGENT);
}

// ---------------------------------------------------------------------------
// Kernel 5: out = relu(out + bias)
// ---------------------------------------------------------------------------
__global__ void k_bias_relu(float* __restrict__ out, const float* __restrict__ bias) {
    int i = blockIdx.x * blockDim.x + threadIdx.x;      // one float4 each
    if (i >= N_NODES * OUT_DIM / 4) return;
    float4 o = ((const float4*)out)[i];
    int col = (i * 4) & (OUT_DIM - 1);                  // 4 | OUT_DIM, no wrap
    float4 b = *(const float4*)(bias + col);
    o.x = fmaxf(o.x + b.x, 0.f);
    o.y = fmaxf(o.y + b.y, 0.f);
    o.z = fmaxf(o.z + b.z, 0.f);
    o.w = fmaxf(o.w + b.w, 0.f);
    ((float4*)out)[i] = o;
}

// ---------------------------------------------------------------------------
extern "C" void kernel_launch(void* const* d_in, const int* in_sizes, int n_in,
                              void* d_out, int out_size, void* d_ws, size_t ws_size,
                              hipStream_t stream) {
    const float* x     = (const float*)d_in[0];
    const float* W     = (const float*)d_in[1];
    const float* bias  = (const float*)d_in[2];
    const int*   esrc  = (const int*)d_in[3];
    const int*   edst  = (const int*)d_in[4];
    const float* evals = (const float*)d_in[5];
    float*  out = (float*)d_out;
    float*  wtd = (float*)d_ws;                                   // 51.2 MB
    __bf16* Wt  = (__bf16*)((char*)d_ws + WS_WT_OFF);             // 128 KB

    const int out4 = N_NODES * OUT_DIM / 4;                       // 3,200,000

    k_wt_convert<<<(IN_DIM * OUT_DIM) / 256, 256, 0, stream>>>(W, Wt);
    k_zero<<<(out4 + 255) / 256, 256, 0, stream>>>((float4*)out, out4);
    k_gemm<<<(N_NODES + 63) / 64, 256, 0, stream>>>(x, Wt, wtd);
    k_spmm<<<N_EDGES / 8, 256, 0, stream>>>(wtd, esrc, edst, evals, out);
    k_bias_relu<<<(out4 + 255) / 256, 256, 0, stream>>>(out, bias);
}
